// MSC_module_15865609191815
// MI455X (gfx1250) — matrix operand order repaired and checked by reading, not compiled, not run on hardware
//
#include <hip/hip_runtime.h>
#include <hip/hip_bf16.h>
#include <stdint.h>

typedef __attribute__((ext_vector_type(16))) __bf16 v16bf;
typedef __attribute__((ext_vector_type(8)))  __bf16 v8bf;
typedef __attribute__((ext_vector_type(8)))  float  v8f;

#define B_    4
#define C_    128
#define N_    4096
#define KC_   8      // 256-dim q/k split into 8 chunks of K=32
#define CT_   8      // 128 channels = 8 tiles of 16
#define NT16  (N_/16)   // 256 row tiles
#define NT32  (N_/32)   // 128 column chunks

// ---- CDNA5 async global->LDS copy (tracked by ASYNCcnt) --------------------
__device__ __forceinline__ void async_cp_b128(uint32_t lds_off, const void* g) {
    asm volatile("global_load_async_to_lds_b128 %0, %1, off"
                 :: "v"(lds_off), "v"((unsigned long long)(uintptr_t)g)
                 : "memory");
}
__device__ __forceinline__ void wait_async0() {
    asm volatile("s_wait_asynccnt 0" ::: "memory");
}

// ---------------------------------------------------------------------------
// Prep: build bf16 Q (A-layout) and K (B-layout) fragment matrices.
// Coalesced global reads into an LDS tile (pad 17 to avoid bank conflicts),
// then permuted fragment emission from LDS.
// A-layout element e of lane L: m = L%16, k = e + 16*(L>=16)
// B-layout element e of lane L: n = L%16, k = e + 16*(L>=16)
// ---------------------------------------------------------------------------
__global__ void prep_qk(const float* __restrict__ P, const float* __restrict__ Q,
                        __bf16* __restrict__ Qmat, __bf16* __restrict__ Kmat) {
    __shared__ float shq[C_][17], shp[C_][17];
    int blk  = blockIdx.x;              // b*NT16 + tile
    int b    = blk / NT16;
    int tile = blk % NT16;
    const float* Pb = P + (size_t)b * C_ * N_ + tile * 16;
    const float* Qb = Q + (size_t)b * C_ * N_ + tile * 16;
    for (int j = threadIdx.x; j < C_ * 16; j += 256) {
        int d = j >> 4, col = j & 15;
        shq[d][col] = Qb[(size_t)d * N_ + col];
        shp[d][col] = Pb[(size_t)d * N_ + col];
    }
    __syncthreads();

    int kc   = threadIdx.x >> 5;
    int lane = threadIdx.x & 31;
    int lh   = lane >> 4;
    int col  = lane & 15;
    v16bf qv, kv;
#pragma unroll
    for (int e = 0; e < 16; ++e) {
        int dA = kc * 32 + e + lh * 16;                      // A-layout k map
        float q = (dA < C_) ? shq[dA][col] : shp[dA - C_][col];
        qv[e] = (__bf16)q;
        int dB = kc * 32 + e + lh * 16;                      // B-layout k map
        float k = (dB < C_) ? (shq[dB][col] + shp[dB][col]) : shp[dB - C_][col];
        kv[e] = (__bf16)k;
    }
    size_t off = ((size_t)(blk * KC_ + kc) * 32 + lane) * 16;
    *(v16bf*)(Qmat + off) = qv;
    *(v16bf*)(Kmat + off) = kv;
}

// V[m,c] = Q[b,c,m] in B-layout per 32-m chunk: lane L -> c = ct*16 + L%16,
// element e -> m_local = e + 16*(L>=16)  (contiguous 64B global reads).
__global__ void prep_v(const float* __restrict__ Q, __bf16* __restrict__ Vmat) {
    int blk  = blockIdx.x;              // b*NT32 + mc
    int ct   = threadIdx.x >> 5;
    int lane = threadIdx.x & 31;
    int b    = blk / NT32;
    int mc   = blk % NT32;
    int c    = ct * 16 + (lane & 15);
    const float* src = Q + ((size_t)b * C_ + c) * N_ + mc * 32 + ((lane < 16) ? 0 : 16);
    v16bf vv;
#pragma unroll
    for (int e = 0; e < 16; ++e) vv[e] = (__bf16)src[e];
    *(v16bf*)(Vmat + ((size_t)(blk * CT_ + ct) * 32 + lane) * 16) = vv;
}

// ---------------------------------------------------------------------------
// Flash attention: 4 waves/block, 1 wave per 16 query rows; all 4 waves share
// double-buffered K/V chunks staged in LDS via async global->LDS DMA.
// 24 v_wmma_f32_16x16x32_bf16 per 32-column chunk per wave.
// ---------------------------------------------------------------------------
__global__ void __launch_bounds__(128) flash(const __bf16* __restrict__ Qmat,
                                             const __bf16* __restrict__ Kmat,
                                             const __bf16* __restrict__ Vmat,
                                             float* __restrict__ Qws) {
    __shared__ __align__(16) __bf16 kbuf[2][2 * KC_ * 512];   // 2 x 16KB
    __shared__ __align__(16) __bf16 vbuf[2][CT_ * 512];       // 2 x 8KB
    __shared__ __align__(16) __bf16 pbuf[4][16][32];          // per-wave probs

    int t    = threadIdx.x;               // 0..127
    int w    = t >> 5;
    int lane = t & 31;
    int lh   = lane >> 4;
    int l16  = lane & 15;
    int gt   = blockIdx.x * 4 + w;        // b*NT16 + tile (b uniform in block)
    int b    = (blockIdx.x * 4) / NT16;
    int tile = gt % NT16;

    uint32_t kb0 = (uint32_t)(uintptr_t)&kbuf[0][0];
    uint32_t vb0 = (uint32_t)(uintptr_t)&vbuf[0][0];

    // Query fragments for this 16-row tile (K=256 -> 8 fragments)
    const __bf16* Qm = Qmat + (size_t)gt * (KC_ * 512) + lane * 16;
    v16bf qf[KC_];
#pragma unroll
    for (int kc = 0; kc < KC_; ++kc)
        qf[kc] = *(const v16bf*)(Qm + kc * 512);

    v8f acc[CT_];
#pragma unroll
    for (int ct = 0; ct < CT_; ++ct)
#pragma unroll
        for (int r = 0; r < 8; ++r) acc[ct][r] = 0.0f;

    float mrow[8], rsum[8];
#pragma unroll
    for (int r = 0; r < 8; ++r) { mrow[r] = -3.0e38f; rsum[r] = 0.0f; }

    // Stage chunk mc into LDS buffer `buf` (24KB, 12 async b128 per thread).
    auto issue_chunk = [&](int mc, int buf) {
        const char* gK = (const char*)(Kmat + ((size_t)(b * NT16 + 2 * mc) * KC_) * 512);
        const char* gV = (const char*)(Vmat + ((size_t)(b * NT32 + mc) * CT_) * 512);
        uint32_t kb = kb0 + (uint32_t)buf * (2 * KC_ * 512 * 2);
        uint32_t vb = vb0 + (uint32_t)buf * (CT_ * 512 * 2);
#pragma unroll
        for (int j = 0; j < 8; ++j)      // 16KB of K fragments
            async_cp_b128(kb + (j * 128 + t) * 16, gK + (j * 128 + t) * 16);
#pragma unroll
        for (int j = 0; j < 4; ++j)      // 8KB of V fragments
            async_cp_b128(vb + (j * 128 + t) * 16, gV + (j * 128 + t) * 16);
    };

    issue_chunk(0, 0);

    for (int mc = 0; mc < NT32; ++mc) {
        int cur = mc & 1;
        wait_async0();        // my share of chunk mc has landed in LDS
        __syncthreads();      // everyone's share landed; prev buffer drained
        if (mc + 1 < NT32) issue_chunk(mc + 1, cur ^ 1);

        const __bf16* kcur = kbuf[cur];
        const __bf16* vcur = vbuf[cur];

        // ---- S tile: 16 rows x 32 cols, K=256 ----
        v8f s0, s1;
#pragma unroll
        for (int r = 0; r < 8; ++r) { s0[r] = 0.0f; s1[r] = 0.0f; }
#pragma unroll
        for (int kc = 0; kc < KC_; ++kc) {
            v16bf bk0 = *(const v16bf*)(kcur + (0 * KC_ + kc) * 512 + lane * 16);
            v16bf bk1 = *(const v16bf*)(kcur + (1 * KC_ + kc) * 512 + lane * 16);
            s0 = __builtin_amdgcn_wmma_f32_16x16x32_bf16(false, qf[kc], false, bk0,
                                                         (short)0, s0, false, false);
            s1 = __builtin_amdgcn_wmma_f32_16x16x32_bf16(false, qf[kc], false, bk1,
                                                         (short)0, s1, false, false);
        }

        // ---- online softmax (row M = r + 8*lh lives in one 16-lane half) ----
        float scl[8];
#pragma unroll
        for (int r = 0; r < 8; ++r) {
            float cm = fmaxf(s0[r], s1[r]);
            cm = fmaxf(cm, __shfl_xor(cm, 1, 16));
            cm = fmaxf(cm, __shfl_xor(cm, 2, 16));
            cm = fmaxf(cm, __shfl_xor(cm, 4, 16));
            cm = fmaxf(cm, __shfl_xor(cm, 8, 16));
            float mnew = fmaxf(mrow[r], cm);
            scl[r] = __expf(mrow[r] - mnew);
            mrow[r] = mnew;
            float p0 = __expf(s0[r] - mnew);
            float p1 = __expf(s1[r] - mnew);
            s0[r] = p0; s1[r] = p1;
            float rs = p0 + p1;
            rs += __shfl_xor(rs, 1, 16);
            rs += __shfl_xor(rs, 2, 16);
            rs += __shfl_xor(rs, 4, 16);
            rs += __shfl_xor(rs, 8, 16);
            rsum[r] = rsum[r] * scl[r] + rs;
        }
#pragma unroll
        for (int ct = 0; ct < CT_; ++ct)
#pragma unroll
            for (int r = 0; r < 8; ++r) acc[ct][r] *= scl[r];

        // ---- C-layout -> A-layout transpose of probs via private LDS ----
#pragma unroll
        for (int r = 0; r < 8; ++r) {
            pbuf[w][r + 8 * lh][l16]      = (__bf16)s0[r];
            pbuf[w][r + 8 * lh][16 + l16] = (__bf16)s1[r];
        }
        const v8bf* rowp = (const v8bf*)&pbuf[w][l16][lh * 16];
        v8bf lo = rowp[0];     // cols base..base+7
        v8bf hi = rowp[1];     // cols base+8..base+15
        v16bf pf;
#pragma unroll
        for (int e = 0; e < 8; ++e) { pf[e] = lo[e]; pf[8 + e] = hi[e]; }

        // ---- acc += P(16x32) @ V(32x16) per channel tile ----
#pragma unroll
        for (int ct = 0; ct < CT_; ++ct) {
            v16bf vf = *(const v16bf*)(vcur + ct * 512 + lane * 16);
            acc[ct] = __builtin_amdgcn_wmma_f32_16x16x32_bf16(false, pf, false, vf,
                                                              (short)0, acc[ct], false, false);
        }
    }

    // ---- epilogue: normalize rows, write Qtile as [b][n][c] (coalesced) ----
#pragma unroll
    for (int r = 0; r < 8; ++r) rsum[r] = 1.0f / rsum[r];
    float* outb = Qws + ((size_t)b * N_ + tile * 16) * C_;
#pragma unroll
    for (int ct = 0; ct < CT_; ++ct)
#pragma unroll
        for (int r = 0; r < 8; ++r) {
            int n = r + 8 * lh;
            outb[(size_t)n * C_ + ct * 16 + l16] = acc[ct][r] * rsum[r];
        }
}

// ---------------------------------------------------------------------------
// Deterministic two-stage per-channel stats (sum, sumsq over B*N rows).
// ---------------------------------------------------------------------------
__global__ void stats1(const float* __restrict__ Qws,
                       float* __restrict__ psum, float* __restrict__ psq) {
    int c  = threadIdx.x & 127;
    int rh = threadIdx.x >> 7;          // 0/1
    int base = blockIdx.x * 128;        // 128 blocks x 128 rows
    float s = 0.0f, s2 = 0.0f;
    for (int r = rh; r < 128; r += 2) {
        float v = Qws[(size_t)(base + r) * C_ + c];
        s += v; s2 += v * v;
    }
    __shared__ float sh[2][128], sh2[2][128];
    sh[rh][c] = s; sh2[rh][c] = s2;
    __syncthreads();
    if (rh == 0) {
        psum[blockIdx.x * C_ + c] = sh[0][c] + sh[1][c];
        psq [blockIdx.x * C_ + c] = sh2[0][c] + sh2[1][c];
    }
}

__global__ void stats2(const float* __restrict__ psum, const float* __restrict__ psq,
                       const float* __restrict__ gamma, const float* __restrict__ bnw,
                       const float* __restrict__ bnb, float* __restrict__ coef) {
    int c = threadIdx.x;                // 128 threads
    float s = 0.0f, s2 = 0.0f;
    for (int j = 0; j < 128; ++j) { s += psum[j * C_ + c]; s2 += psq[j * C_ + c]; }
    const float inv = 1.0f / (float)(B_ * N_);
    float mq = s * inv;
    float vq = s2 * inv - mq * mq;      // biased variance of Qtile
    float g  = gamma[0];
    float rstd = rsqrtf(g * g * vq + 1e-5f);   // y = g*q -> var_y = g^2 var_q
    coef[c]      = g * rstd * bnw[c];                   // scale
    coef[C_ + c] = -g * mq * rstd * bnw[c] + bnb[c];    // shift
}

// ---------------------------------------------------------------------------
// Finalize: [n][c] -> [c][n] LDS transpose + BN affine + ReLU + residual.
// ---------------------------------------------------------------------------
__global__ void finalize(const float* __restrict__ Qws, const float* __restrict__ coef,
                         const float* __restrict__ xout, float* __restrict__ out) {
    __shared__ float tl[16][132];
    int blk = blockIdx.x;               // b*NT16 + tile
    int b = blk / NT16, t16 = blk % NT16;
    const float* src = Qws + ((size_t)b * N_ + t16 * 16) * C_;
    for (int k = threadIdx.x; k < 16 * C_; k += 256) {
        int row = k >> 7, c = k & 127;
        tl[row][c] = src[(size_t)row * C_ + c];
    }
    __syncthreads();
    int c   = threadIdx.x >> 1;
    int nn0 = (threadIdx.x & 1) * 8;
    float a = coef[c], bb = coef[C_ + c];
    size_t obase = ((size_t)b * C_ + c) * N_ + t16 * 16 + nn0;
#pragma unroll
    for (int j = 0; j < 8; ++j) {
        float y = fmaxf(a * tl[nn0 + j][c] + bb, 0.0f);
        out[obase + j] = y + xout[obase + j];
    }
}

// ---------------------------------------------------------------------------
extern "C" void kernel_launch(void* const* d_in, const int* in_sizes, int n_in,
                              void* d_out, int out_size, void* d_ws, size_t ws_size,
                              hipStream_t stream) {
    const float* x_in  = (const float*)d_in[0];   // P
    const float* x_out = (const float*)d_in[1];   // Q
    const float* gamma = (const float*)d_in[2];
    const float* bnw   = (const float*)d_in[3];
    const float* bnb   = (const float*)d_in[4];
    float* out = (float*)d_out;

    char* ws = (char*)d_ws;
    __bf16* Qmat = (__bf16*)(ws);                      //  8 MB
    __bf16* Kmat = (__bf16*)(ws + ((size_t)8  << 20)); //  8 MB
    __bf16* Vmat = (__bf16*)(ws + ((size_t)16 << 20)); //  4 MB
    float*  Qws  = (float* )(ws + ((size_t)20 << 20)); //  8 MB
    float*  psum = (float* )(ws + ((size_t)28 << 20)); // 64 KB
    float*  psq  = psum + 128 * C_;                    // 64 KB
    float*  coef = psq  + 128 * C_;                    //  1 KB

    prep_qk <<<B_ * NT16,     256, 0, stream>>>(x_in, x_out, Qmat, Kmat);
    prep_v  <<<B_ * NT32,     256, 0, stream>>>(x_out, Vmat);
    flash   <<<B_ * NT16 / 4, 128, 0, stream>>>(Qmat, Kmat, Vmat, Qws);
    stats1  <<<128,           256, 0, stream>>>(Qws, psum, psq);
    stats2  <<<1,             128, 0, stream>>>(psum, psq, gamma, bnw, bnb, coef);
    finalize<<<B_ * NT16,     256, 0, stream>>>(Qws, coef, x_out, out);
}
